// GATLayer_23364622090806
// MI455X (gfx1250) — compile-verified, matrix operand order checked by reading, not run on hardware
//
#include <hip/hip_runtime.h>
#include <hip/hip_bf16.h>

typedef __attribute__((ext_vector_type(16))) _Float16 v16h;
typedef __attribute__((ext_vector_type(8)))  float    v8f;

#define N_NODES  100000
#define E_EDGES  1600000
#define IN_NDIM  128
#define OUT_NDIM 64

// ---- ordered-int encoding of float for atomicMax ----
__device__ __forceinline__ int f2ord(float f) {
  int i = __float_as_int(f);
  return (i >= 0) ? i : (i ^ 0x7FFFFFFF);
}
__device__ __forceinline__ float ord2f(int i) {
  return __int_as_float((i >= 0) ? i : (i ^ 0x7FFFFFFF));
}
// f2ord(-inf) == 0x807FFFFF
#define EMAX_SENTINEL ((int)0x807FFFFF)

// ---------------------------------------------------------------------------
// Kernel 1: z = h @ W1, z_i = h @ W2 via v_wmma_f32_16x16x32_f16.
// 256 threads = 8 wave32 waves. Block handles a 16-row tile of h.
// Waves 0-3 -> z col tiles 0..3; waves 4-7 -> z_i col tiles 0..3.
// K-loop fully unrolled: 4 back-to-back WMMAs, A loads vectorized as b128.
// ---------------------------------------------------------------------------
__global__ __launch_bounds__(256) void gat_node_gemm(
    const float* __restrict__ h, const float* __restrict__ W1,
    const float* __restrict__ W2, float* __restrict__ z,
    float* __restrict__ zi)
{
  const int wave = threadIdx.x >> 5;
  const int lane = threadIdx.x & 31;
  const int half = lane >> 4;   // which 16-lane half of the wave
  const int lm   = lane & 15;   // A: row m within tile; B: col n within tile
  const int row0 = blockIdx.x * 16;

  const float* __restrict__ W   = (wave < 4) ? W1 : W2;
  float*       __restrict__ out = (wave < 4) ? z  : zi;
  const int nc = (wave & 3) * 16;

  v8f acc = {};
  const float* __restrict__ arow = h + (size_t)(row0 + lm) * IN_NDIM;

#pragma unroll
  for (int kb = 0; kb < IN_NDIM; kb += 32) {
    // A fragment: 16x32 f16 (ISA 16-bit A layout).
    // Lane(lm,half): elems 0..7 -> K = half*8 + j ; elems 8..15 -> K = 16+half*8+j
    // 8 consecutive floats each -> two float4 (global_load_b128) per group.
    const float4 a0 = *(const float4*)(arow + kb + half * 8);
    const float4 a1 = *(const float4*)(arow + kb + half * 8 + 4);
    const float4 a2 = *(const float4*)(arow + kb + 16 + half * 8);
    const float4 a3 = *(const float4*)(arow + kb + 16 + half * 8 + 4);

    v16h a, b;
    a[0]  = (_Float16)a0.x;  a[1]  = (_Float16)a0.y;
    a[2]  = (_Float16)a0.z;  a[3]  = (_Float16)a0.w;
    a[4]  = (_Float16)a1.x;  a[5]  = (_Float16)a1.y;
    a[6]  = (_Float16)a1.z;  a[7]  = (_Float16)a1.w;
    a[8]  = (_Float16)a2.x;  a[9]  = (_Float16)a2.y;
    a[10] = (_Float16)a2.z;  a[11] = (_Float16)a2.w;
    a[12] = (_Float16)a3.x;  a[13] = (_Float16)a3.y;
    a[14] = (_Float16)a3.z;  a[15] = (_Float16)a3.w;

    // B fragment: 32x16 f16, mirrored K mapping; lane holds column nc+lm.
    // Stride-64 reads; W1/W2 are 32 KB and stay cache-resident.
#pragma unroll
    for (int j = 0; j < 8; ++j) {
      b[j]     = (_Float16)W[(size_t)(kb + half * 8 + j) * OUT_NDIM + nc + lm];
      b[j + 8] = (_Float16)W[(size_t)(kb + 16 + half * 8 + j) * OUT_NDIM + nc + lm];
    }

    acc = __builtin_amdgcn_wmma_f32_16x16x32_f16(
        /*neg_a=*/false, a, /*neg_b=*/false, b,
        /*c_mod=*/(short)0, acc, /*reuse_a=*/false, /*reuse_b=*/false);
  }

  // D layout: VGPR r, lane -> (m = r + half*8, n = lm)
#pragma unroll
  for (int r = 0; r < 8; ++r) {
    out[(size_t)(row0 + r + half * 8) * OUT_NDIM + nc + lm] = acc[r];
  }
}

// ---------------------------------------------------------------------------
// Kernel 2: init accumulators (out=0, denom=0, emax=-inf sentinel)
// ---------------------------------------------------------------------------
__global__ void gat_init(float* __restrict__ out, float* __restrict__ denom,
                         int* __restrict__ emax)
{
  int i = blockIdx.x * blockDim.x + threadIdx.x;
  if (i < N_NODES * OUT_NDIM) out[i] = 0.0f;
  if (i < N_NODES) {
    denom[i] = 0.0f;
    emax[i]  = EMAX_SENTINEL;
  }
}

// ---------------------------------------------------------------------------
// Kernel 3: per-node attention scalars
//   a_src[n] = z[n] . Wa[0:64],  a_dst[n] = z[n] . Wa[64:128]
// ---------------------------------------------------------------------------
__global__ void gat_attn_node(const float* __restrict__ z,
                              const float* __restrict__ Wa,
                              float* __restrict__ asrc,
                              float* __restrict__ adst)
{
  int i = blockIdx.x * blockDim.x + threadIdx.x;
  if (i >= N_NODES) return;
  const float* __restrict__ zr = z + (size_t)i * OUT_NDIM;
  float s1 = 0.0f, s2 = 0.0f;
#pragma unroll 8
  for (int k = 0; k < OUT_NDIM; ++k) {
    float zv = zr[k];
    s1 += zv * Wa[k];
    s2 += zv * Wa[OUT_NDIM + k];
  }
  asrc[i] = s1;
  adst[i] = s2;
}

// ---------------------------------------------------------------------------
// Kernel 4: edge scores e = leaky_relu(a_src[src] + a_dst[dst] + d*W0*Wa[128])
//           + per-dst running max (ordered-int atomicMax)
// ---------------------------------------------------------------------------
__global__ void gat_edge_score(const float* __restrict__ d,
                               const float* __restrict__ W0,
                               const float* __restrict__ Wa,
                               const int* __restrict__ src,
                               const int* __restrict__ dst,
                               const float* __restrict__ asrc,
                               const float* __restrict__ adst,
                               float* __restrict__ ebuf,
                               int* __restrict__ emax)
{
  int e = blockIdx.x * blockDim.x + threadIdx.x;
  if (e >= E_EDGES) return;
  const float c = W0[0] * Wa[2 * OUT_NDIM];  // Wa row 128
  int dd = dst[e];
  float ev = asrc[src[e]] + adst[dd] + d[e] * c;
  ev = (ev > 0.0f) ? ev : 0.01f * ev;  // leaky_relu(0.01)
  ebuf[e] = ev;
  atomicMax(&emax[dd], f2ord(ev));
}

// ---------------------------------------------------------------------------
// Kernel 5: ex = exp(e - emax[dst]); denom[dst] += ex
// ---------------------------------------------------------------------------
__global__ void gat_edge_exp(const int* __restrict__ dst,
                             float* __restrict__ ebuf,
                             const int* __restrict__ emax,
                             float* __restrict__ denom)
{
  int e = blockIdx.x * blockDim.x + threadIdx.x;
  if (e >= E_EDGES) return;
  int dd = dst[e];
  float ex = __expf(ebuf[e] - ord2f(emax[dd]));
  ebuf[e] = ex;
  atomicAdd(&denom[dd], ex);
}

// ---------------------------------------------------------------------------
// Kernel 6: weighted scatter — one thread per (edge, feature)
//   out[dst] += (ex/denom[dst]) * z[src]
// ---------------------------------------------------------------------------
__global__ __launch_bounds__(256) void gat_scatter(
    const int* __restrict__ src, const int* __restrict__ dst,
    const float* __restrict__ ebuf, const float* __restrict__ denom,
    const float* __restrict__ z, float* __restrict__ out)
{
  int idx = blockIdx.x * 256 + threadIdx.x;  // E*64 = 102.4M < 2^31
  int e = idx >> 6;
  int k = idx & 63;
  int s  = src[e];
  int dd = dst[e];
  float alpha = ebuf[e] / denom[dd];
  atomicAdd(&out[(size_t)dd * OUT_NDIM + k],
            alpha * z[(size_t)s * OUT_NDIM + k]);
}

// ---------------------------------------------------------------------------
// Kernel 7: out = relu(z_i + out)
// ---------------------------------------------------------------------------
__global__ void gat_final(const float* __restrict__ zi, float* __restrict__ out)
{
  int i = blockIdx.x * blockDim.x + threadIdx.x;
  if (i >= N_NODES * OUT_NDIM) return;
  float v = zi[i] + out[i];
  out[i] = (v > 0.0f) ? v : 0.0f;
}

// ---------------------------------------------------------------------------
extern "C" void kernel_launch(void* const* d_in, const int* in_sizes, int n_in,
                              void* d_out, int out_size, void* d_ws, size_t ws_size,
                              hipStream_t stream)
{
  // setup_inputs order: h, d, W0, W1, W2, Wa, src, dst
  const float* h   = (const float*)d_in[0];
  const float* dfe = (const float*)d_in[1];
  const float* W0  = (const float*)d_in[2];
  const float* W1  = (const float*)d_in[3];
  const float* W2  = (const float*)d_in[4];
  const float* Wa  = (const float*)d_in[5];
  const int*   src = (const int*)d_in[6];
  const int*   dst = (const int*)d_in[7];
  float* out = (float*)d_out;

  // workspace carve-up
  char* ws = (char*)d_ws;
  float* z     = (float*)ws;  ws += (size_t)N_NODES * OUT_NDIM * sizeof(float);
  float* zi    = (float*)ws;  ws += (size_t)N_NODES * OUT_NDIM * sizeof(float);
  float* ebuf  = (float*)ws;  ws += (size_t)E_EDGES * sizeof(float);
  float* asrc  = (float*)ws;  ws += (size_t)N_NODES * sizeof(float);
  float* adst  = (float*)ws;  ws += (size_t)N_NODES * sizeof(float);
  float* denom = (float*)ws;  ws += (size_t)N_NODES * sizeof(float);
  int*   emax  = (int*)ws;    ws += (size_t)N_NODES * sizeof(int);

  const int TB = 256;
  // 1) WMMA node GEMMs (N divides 16 exactly: 6250 row tiles)
  gat_node_gemm<<<N_NODES / 16, TB, 0, stream>>>(h, W1, W2, z, zi);
  // 2) init accumulators (independent of GEMM)
  gat_init<<<(N_NODES * OUT_NDIM + TB - 1) / TB, TB, 0, stream>>>(out, denom, emax);
  // 3) per-node attention scalars
  gat_attn_node<<<(N_NODES + TB - 1) / TB, TB, 0, stream>>>(z, Wa, asrc, adst);
  // 4) edge scores + segment max
  gat_edge_score<<<E_EDGES / TB, TB, 0, stream>>>(dfe, W0, Wa, src, dst,
                                                  asrc, adst, ebuf, emax);
  // 5) exp + segment sum
  gat_edge_exp<<<E_EDGES / TB, TB, 0, stream>>>(dst, ebuf, emax, denom);
  // 6) weighted neighbor aggregation into out
  gat_scatter<<<(E_EDGES * OUT_NDIM) / TB, TB, 0, stream>>>(src, dst, ebuf,
                                                            denom, z, out);
  // 7) relu(z_i + out)
  gat_final<<<(N_NODES * OUT_NDIM + TB - 1) / TB, TB, 0, stream>>>(zi, out);
}
